// TemporalSum_29892972380509
// MI455X (gfx1250) — matrix operand order repaired and checked by reading, not compiled, not run on hardware
//
#include <hip/hip_runtime.h>

typedef __attribute__((ext_vector_type(16))) _Float16 v16h;
typedef __attribute__((ext_vector_type(8)))  _Float16 v8h;
typedef __attribute__((ext_vector_type(8)))  float    v8f;
typedef __attribute__((ext_vector_type(4)))  int      v4i;

#define EMB 128
#define EF  128
#define TILE_M 128
#define THREADS 256

// ---- gfx1250 async global->LDS support (probe via __has_builtin) -----------
#if defined(__has_builtin)
#if __has_builtin(__builtin_amdgcn_global_load_async_to_lds_b128)
#define HAVE_ASYNC_LDS 1
#endif
#endif
#ifndef HAVE_ASYNC_LDS
#define HAVE_ASYNC_LDS 0
#endif

// Builtin wants: (v4i addrspace(1)*, v4i addrspace(3)*, imm offset, imm cpol)
typedef __attribute__((address_space(1))) v4i* g_v4i_ptr;
typedef __attribute__((address_space(3))) v4i* l_v4i_ptr;

__device__ __forceinline__ void wait_asynccnt0() {
#if defined(__has_builtin)
#if __has_builtin(__builtin_amdgcn_s_wait_asynccnt)
  __builtin_amdgcn_s_wait_asynccnt(0);
  return;
#else
  asm volatile("s_wait_asynccnt 0" ::: "memory");
  return;
#endif
#else
  asm volatile("s_wait_asynccnt 0" ::: "memory");
#endif
}

// LDS layout (in _Float16 elements):
//   sW1  : 128*256   W1  (f16)           sW2  : 128*128   W2  (f16)
//   smW1 : 128*256   mW1 (f16)           smW2 : 128*128   mW2 (f16)
//   sC   : 128*256   C tile / X tile (also f32 weight staging at startup)
//   sT   : 128*128   intermediate
//   then float[4*128] biases (b1,b2,mb1,mb2) + float[256] (time_w,time_b)
#define OFF_W1   0
#define OFF_W2   (OFF_W1  + 128*256)
#define OFF_MW1  (OFF_W2  + 128*128)
#define OFF_MW2  (OFF_MW1 + 128*256)
#define OFF_C    (OFF_MW2 + 128*128)
#define OFF_T    (OFF_C   + 128*256)
#define OFF_END  (OFF_T   + 128*128)
#define SMEM_BYTES (OFF_END * 2 + (4*128 + 256) * 4)

#define CHUNK_F32 16384   // 64KB staging chunk (fits in sC)

// ---- WMMA fragment loads from LDS (row-major storage) ----------------------
// A 16x32 f16 (ISA 7.12.2): lane l -> row M = l&15; lanes 0-15 hold K 0-7 &
// 16-23, lanes 16-31 hold K 8-15 & 24-31. Two 16B LDS loads per fragment.
__device__ __forceinline__ v16h load_frag_A(const _Float16* base, int lane,
                                            int lda, int k0) {
  int m  = lane & 15;
  int kh = (lane >> 4) * 8;
  union { v16h v; v8h h[2]; } u;
  const _Float16* p = base + m * lda + k0 + kh;
  u.h[0] = *(const v8h*)(p);
  u.h[1] = *(const v8h*)(p + 16);
  return u.v;
}

// B 32x16 f16: B[k][n] = W[n0+n][k] with W row-major [out][in].
// lane l -> col N = l&15; lanes 0-15 hold K 0-7 & 16-23, lanes 16-31 hold K 8-15 & 24-31.
__device__ __forceinline__ v16h load_frag_B(const _Float16* wbase, int lane,
                                            int ldw, int n0, int k0) {
  int n  = lane & 15;
  int kh = (lane >> 4) * 8;
  union { v16h v; v8h h[2]; } u;
  const _Float16* p = wbase + (n0 + n) * ldw + k0 + kh;
  u.h[0] = *(const v8h*)(p);
  u.h[1] = *(const v8h*)(p + 16);
  return u.v;
}

// One 16-row x 128-col GEMM strip: acc[8] n-tiles (independent WMMA chains
// for ILP on the XDL pipe), ksteps*32 deep.
__device__ __forceinline__ void gemm_strip(const _Float16* Abase, int lda,
                                           int ksteps, const _Float16* W,
                                           int ldw, int lane, v8f acc[8]) {
#pragma unroll
  for (int n = 0; n < 8; ++n)
#pragma unroll
    for (int i = 0; i < 8; ++i) acc[n][i] = 0.0f;
#pragma unroll
  for (int k = 0; k < 8; ++k) {
    if (k >= ksteps) break;
    v16h a = load_frag_A(Abase, lane, lda, k * 32);
#pragma unroll
    for (int n = 0; n < 8; ++n) {
      v16h b = load_frag_B(W, lane, ldw, n * 16, k * 32);
      acc[n] = __builtin_amdgcn_wmma_f32_16x16x32_f16(
          false, a, false, b, (short)0, acc[n], false, false);
    }
  }
}

// Stage a f32 weight matrix into LDS as f16. When available, use gfx1250
// async global->LDS copies (ASYNCcnt path, no VGPR round-trip for the raw
// data) through a 64KB f32 staging region, then convert LDS->LDS.
__device__ __forceinline__ void load_weight_f16(const float* __restrict__ gsrc,
                                                _Float16* dst, int nElem,
                                                float* stage, int tid) {
#if HAVE_ASYNC_LDS
  for (int base = 0; base < nElem; base += CHUNK_F32) {
    for (int i = tid * 4; i < CHUNK_F32; i += THREADS * 4) {
      __builtin_amdgcn_global_load_async_to_lds_b128(
          (g_v4i_ptr)(gsrc + base + i), (l_v4i_ptr)(stage + i), 0, 0);
    }
    wait_asynccnt0();
    __syncthreads();
    for (int i = tid * 4; i < CHUNK_F32; i += THREADS * 4) {
      float4 v = *(const float4*)(stage + i);
      _Float16* d = dst + base + i;
      d[0] = (_Float16)v.x; d[1] = (_Float16)v.y;
      d[2] = (_Float16)v.z; d[3] = (_Float16)v.w;
    }
    __syncthreads();
  }
#else
  (void)stage;
  for (int i = tid * 4; i < nElem; i += THREADS * 4) {
    float4 v = *(const float4*)(gsrc + i);
    _Float16* d = dst + i;
    d[0] = (_Float16)v.x; d[1] = (_Float16)v.y;
    d[2] = (_Float16)v.z; d[3] = (_Float16)v.w;
  }
#endif
}

extern "C" __global__ __launch_bounds__(THREADS)
void tgn_layer_kernel(const float* __restrict__ h_in,
                      const float* __restrict__ feat,
                      const float* __restrict__ timestamp,
                      const float* __restrict__ last_update,
                      const float* __restrict__ time_w,
                      const float* __restrict__ time_b,
                      const float* __restrict__ W1, const float* __restrict__ b1,
                      const float* __restrict__ W2, const float* __restrict__ b2,
                      const float* __restrict__ mW1, const float* __restrict__ mb1,
                      const float* __restrict__ mW2, const float* __restrict__ mb2,
                      const int* __restrict__ src,
                      float* __restrict__ h_out,
                      int n_nodes) {
  extern __shared__ __align__(16) _Float16 smem[];
  _Float16* sW1  = smem + OFF_W1;
  _Float16* sW2  = smem + OFF_W2;
  _Float16* smW1 = smem + OFF_MW1;
  _Float16* smW2 = smem + OFF_MW2;
  _Float16* sC   = smem + OFF_C;
  _Float16* sT   = smem + OFF_T;
  float* sBias = (float*)(smem + OFF_END);   // [0:128)=b1 [128:256)=b2
                                             // [256:384)=mb1 [384:512)=mb2
  float* sTW = sBias + 512;                  // [0:128)=time_w [128:256)=time_b
  float* stage = (float*)sC;                 // 64KB f32 staging (startup only)

  const int tid  = threadIdx.x;
  const int lane = tid & 31;
  const int wave = tid >> 5;        // 0..7
  const int mrow = wave * 16;       // wave-local row block within tile

  // ---- weights -> LDS f16 (async global->LDS staging when available) ------
  load_weight_f16(W1,  sW1,  128 * 256, stage, tid);
  load_weight_f16(W2,  sW2,  128 * 128, stage, tid);
  load_weight_f16(mW1, smW1, 128 * 256, stage, tid);
  load_weight_f16(mW2, smW2, 128 * 128, stage, tid);
  if (tid < 128) {
    sBias[tid]       = b1[tid];
    sBias[128 + tid] = b2[tid];
    sBias[256 + tid] = mb1[tid];
    sBias[384 + tid] = mb2[tid];
    sTW[tid]         = time_w[tid];
    sTW[128 + tid]   = time_b[tid];
  }
  __syncthreads();

  // ---- ingest C tile: [ h[src] | feat + cos(dt*w + b) ], f16 in LDS --------
  const int row_base = blockIdx.x * TILE_M;
  {
    int r    = tid >> 1;                 // row within tile, 0..127
    int side = tid & 1;                  // 0 = left half, 1 = right half
    int e = row_base + r;
    if (e >= n_nodes) e = n_nodes - 1;   // clamp (stores guarded later)
    int s = src[e];
    if (side == 0) {
      const float4* hp = (const float4*)(h_in + (size_t)s * EMB);
#pragma unroll
      for (int j = 0; j < EMB / 4; ++j) {
        float4 v = hp[j];
        int c = 4 * j;
        sC[r*256 + c+0] = (_Float16)v.x; sC[r*256 + c+1] = (_Float16)v.y;
        sC[r*256 + c+2] = (_Float16)v.z; sC[r*256 + c+3] = (_Float16)v.w;
      }
    } else {
      float dtv = timestamp[e] - last_update[s];
      const float4* fp = (const float4*)(feat + (size_t)e * EF);
#pragma unroll
      for (int j = 0; j < EF / 4; ++j) {
        float4 fv = fp[j];
        int c = 4 * j;
        float f[4] = {fv.x, fv.y, fv.z, fv.w};
#pragma unroll
        for (int q = 0; q < 4; ++q) {
          float ang = dtv * sTW[c + q] + sTW[128 + c + q];
          // v_cos_f32 takes revolutions: pre-scale by 1/(2*pi)
          float te  = __builtin_amdgcn_cosf(ang * 0.15915494309189535f);
          sC[r*256 + 128 + c + q] = (_Float16)(f[q] + te);
        }
      }
    }
  }
  __syncthreads();

  v8f acc[8];
  const int mh = (lane >> 4) * 8;   // C/D layout: M = v + 8*(lane>>4)
  const int nc = lane & 15;         // C/D layout: N = lane&15

  // ---- GEMM1: C[128,256] @ W1^T -> +b1 -> sT (f16) -------------------------
  gemm_strip(sC + mrow * 256, 256, 8, sW1, 256, lane, acc);
#pragma unroll
  for (int n = 0; n < 8; ++n) {
    float bias = sBias[n * 16 + nc];
#pragma unroll
    for (int v = 0; v < 8; ++v)
      sT[(mrow + mh + v) * 128 + n * 16 + nc] = (_Float16)(acc[n][v] + bias);
  }

  // ---- GEMM2: sT @ W2^T -> +b2, ReLU -> right half of X (sC) ---------------
  gemm_strip(sT + mrow * 128, 128, 4, sW2, 128, lane, acc);
#pragma unroll
  for (int n = 0; n < 8; ++n) {
    float bias = sBias[128 + n * 16 + nc];
#pragma unroll
    for (int v = 0; v < 8; ++v) {
      float val = acc[n][v] + bias;
      val = val > 0.0f ? val : 0.0f;
      sC[(mrow + mh + v) * 256 + 128 + n * 16 + nc] = (_Float16)val;
    }
  }

  // ---- refill left half of X with h[node] (dst == arange) ------------------
  {
    int r = mrow + (lane >> 1);          // 2 lanes per row, 64 cols each
    int node = row_base + r;
    if (node >= n_nodes) node = n_nodes - 1;
    int coff = (lane & 1) * 64;
    const float4* hp = (const float4*)(h_in + (size_t)node * EMB + coff);
#pragma unroll
    for (int j = 0; j < 16; ++j) {
      float4 v = hp[j];
      int c = coff + 4 * j;
      sC[r*256 + c+0] = (_Float16)v.x; sC[r*256 + c+1] = (_Float16)v.y;
      sC[r*256 + c+2] = (_Float16)v.z; sC[r*256 + c+3] = (_Float16)v.w;
    }
  }

  // ---- GEMM3: X[128,256] @ mW1^T -> +mb1, ReLU -> sT (f16) -----------------
  gemm_strip(sC + mrow * 256, 256, 8, smW1, 256, lane, acc);
#pragma unroll
  for (int n = 0; n < 8; ++n) {
    float bias = sBias[256 + n * 16 + nc];
#pragma unroll
    for (int v = 0; v < 8; ++v) {
      float val = acc[n][v] + bias;
      val = val > 0.0f ? val : 0.0f;
      sT[(mrow + mh + v) * 128 + n * 16 + nc] = (_Float16)val;
    }
  }

  // ---- GEMM4: sT @ mW2^T -> +mb2 -> h_out (f32, global) --------------------
  gemm_strip(sT + mrow * 128, 128, 4, smW2, 128, lane, acc);
#pragma unroll
  for (int n = 0; n < 8; ++n) {
    float bias = sBias[384 + n * 16 + nc];
#pragma unroll
    for (int v = 0; v < 8; ++v) {
      int row = row_base + mrow + mh + v;
      if (row < n_nodes)
        h_out[(size_t)row * EMB + n * 16 + nc] = acc[n][v] + bias;
    }
  }
}

extern "C" void kernel_launch(void* const* d_in, const int* in_sizes, int n_in,
                              void* d_out, int out_size, void* d_ws, size_t ws_size,
                              hipStream_t stream) {
  const float* h    = (const float*)d_in[0];
  const float* feat = (const float*)d_in[1];
  const float* ts   = (const float*)d_in[2];
  const float* lu   = (const float*)d_in[3];
  const float* tw   = (const float*)d_in[4];
  const float* tb   = (const float*)d_in[5];
  const float* W1   = (const float*)d_in[6];
  const float* b1   = (const float*)d_in[7];
  const float* W2   = (const float*)d_in[8];
  const float* b2   = (const float*)d_in[9];
  const float* mW1  = (const float*)d_in[10];
  const float* mb1  = (const float*)d_in[11];
  const float* mW2  = (const float*)d_in[12];
  const float* mb2  = (const float*)d_in[13];
  const int*   src  = (const int*)d_in[14];

  const int n_nodes = in_sizes[3];              // last_update has N elements
  float* h_mid = (float*)d_ws;                  // layer-1 output (N*EMB f32)
  float* out   = (float*)d_out;

  const int grid = (n_nodes + TILE_M - 1) / TILE_M;

  // Layer 1: h -> h_mid ; Layer 2: h_mid -> out
  tgn_layer_kernel<<<grid, THREADS, SMEM_BYTES, stream>>>(
      h, feat, ts, lu, tw, tb, W1, b1, W2, b2, mW1, mb1, mW2, mb2, src,
      h_mid, n_nodes);
  tgn_layer_kernel<<<grid, THREADS, SMEM_BYTES, stream>>>(
      h_mid, feat, ts, lu, tw, tb, W1, b1, W2, b2, mW1, mb1, mW2, mb2, src,
      out, n_nodes);
}
